// SpikingMultiHeadAttention_59957743452585
// MI455X (gfx1250) — matrix operand order repaired and checked by reading, not compiled, not run on hardware
//
#include <hip/hip_runtime.h>
#include <hip/hip_bf16.h>

// ---------------------------------------------------------------------------
// MI455X (gfx1250) feature-axis MHA block.
// One generic bf16-WMMA GEMM (v_wmma_f32_16x16x32_bf16, wave32) drives all
// six GEMMs. 128x256x32 block tile, 8 waves, 64x64 per wave (16 WMMA / 16
// ds_load_b128 per BK step), double-buffered LDS with register prefetch,
// f32 accumulation, fused alpha+bias epilogue. Softmax is a row kernel.
// ---------------------------------------------------------------------------

typedef __bf16 bf16_t;
typedef __attribute__((ext_vector_type(16))) __bf16 v16bf;
typedef __attribute__((ext_vector_type(8)))  float  v8f;

#define BM 128
#define BN 256
#define BK 32
#define LDT (BK + 8)          // padded LDS row (bf16 elems) to spread banks

union FragAB {                // one WMMA A/B operand: 16 bf16 per lane
    v16bf v;
    uint4 u[2];
};

// --- staging helpers: global -> packed-bf16 regs -> LDS ---------------------

__device__ inline uint2 pack4(const float4& x) {
    union { bf16_t h[4]; uint2 u; } t;
    t.h[0] = (bf16_t)x.x; t.h[1] = (bf16_t)x.y;
    t.h[2] = (bf16_t)x.z; t.h[3] = (bf16_t)x.w;
    return t.u;
}

// Load a [D0 x BK] tile (D0 = BM or BN) into packed bf16 registers,
// converting f32 on the fly. K_CONTIG picks the unit-stride global axis so
// every load is a coalesced 8/16-byte vector load.
template <typename T, int D0, bool K_CONTIG, int REPS>
__device__ inline void stage_gload(const T* __restrict__ g,
                                   long long stride_d0, long long stride_k,
                                   int tid, uint2 (&r)[REPS]) {
#pragma unroll
    for (int rep = 0; rep < REPS; ++rep) {
        const int s = rep * 256 + tid;
        const T* p;
        if constexpr (K_CONTIG) {            // stride_k == 1
            const int row = s >> 3, k0 = (s & 7) * 4;
            p = g + (long long)row * stride_d0 + k0;
        } else {                             // stride_d0 == 1
            constexpr int G = D0 / 4;
            const int k = s / G, d0 = (s % G) * 4;
            p = g + (long long)k * stride_k + d0;
        }
        if constexpr (sizeof(T) == 4) r[rep] = pack4(*(const float4*)p);
        else                          r[rep] = *(const uint2*)p;
    }
}

template <int D0, bool K_CONTIG, int REPS>
__device__ inline void stage_sstore(bf16_t* __restrict__ sT, int tid,
                                    const uint2 (&r)[REPS]) {
#pragma unroll
    for (int rep = 0; rep < REPS; ++rep) {
        const int s = rep * 256 + tid;
        if constexpr (K_CONTIG) {
            const int row = s >> 3, k0 = (s & 7) * 4;
            *(uint2*)(sT + row * LDT + k0) = r[rep];
        } else {
            constexpr int G = D0 / 4;
            const int k = s / G, d0 = (s % G) * 4;
            union { bf16_t h[4]; uint2 u; } t; t.u = r[rep];
#pragma unroll
            for (int i = 0; i < 4; ++i) sT[(d0 + i) * LDT + k] = t.h[i];
        }
    }
}

// --- generic batched GEMM ---------------------------------------------------
// C[bz,m,n] = alpha * sum_k A[m,k]*B[k,n] + bias[n]
// A addressed as A + bz*batchA + m*a_rs + k*a_cs (same pattern for B).
// Requires M%128==0, N%256==0, K%32==0 (true for every call here).
template <typename AT, typename BT, bool AKC, bool BKC, bool OUT_BF16>
__global__ __launch_bounds__(256, 1)
void gemm_wmma(const AT* __restrict__ A, const BT* __restrict__ B,
               void* __restrict__ C, const float* __restrict__ bias,
               int K,
               long long a_rs, long long a_cs,
               long long b_rs, long long b_cs,
               long long batchA, long long batchB, long long batchC,
               int ldc, float alpha) {
    __shared__ __align__(16) bf16_t sA[2][BM * LDT];
    __shared__ __align__(16) bf16_t sB[2][BN * LDT];

    const int tid  = threadIdx.x;
    const int wave = tid >> 5;        // 8 waves
    const int lane = tid & 31;
    const int wm   = wave >> 2;       // 0..1 -> 64-row strip
    const int wn   = wave & 3;        // 0..3 -> 64-col strip
    const int bz   = blockIdx.z;
    const long long m0 = (long long)blockIdx.y * BM;
    const long long n0 = (long long)blockIdx.x * BN;

    const AT* Ab = A + (long long)bz * batchA + m0 * a_rs;
    const BT* Bb = B + (long long)bz * batchB + n0 * b_cs;

    constexpr int AR = BM / 32;       // staging reps (4)
    constexpr int BR = BN / 32;       // staging reps (8)
    uint2 ar[AR], br[BR];

    v8f acc[4][4];
#pragma unroll
    for (int i = 0; i < 4; ++i)
#pragma unroll
        for (int j = 0; j < 4; ++j)
            acc[i][j] = (v8f)(0.0f);

    const int l16   = lane & 15;
    const int halfA = (lane >> 4) * 8;    // A frag: K 0-7 vs 8-15 halves
    const int halfB = (lane >> 4) * 8;    // B frag: K 0-7 vs 8-15 halves

    const int KT = K / BK;
    // prologue: tile 0 into registers
    stage_gload<AT, BM, AKC>(Ab, a_rs, a_cs, tid, ar);
    stage_gload<BT, BN, BKC>(Bb, b_cs, b_rs, tid, br);

    int buf = 0;
    for (int kt = 0; kt < KT; ++kt) {
        // commit tile kt to LDS[buf]
        stage_sstore<BM, AKC>(sA[buf], tid, ar);
        stage_sstore<BN, BKC>(sB[buf], tid, br);
        __syncthreads();

        // prefetch tile kt+1 into registers while WMMAs run on LDS[buf]
        if (kt + 1 < KT) {
            stage_gload<AT, BM, AKC>(Ab + (long long)(kt + 1) * BK * a_cs,
                                     a_rs, a_cs, tid, ar);
            stage_gload<BT, BN, BKC>(Bb + (long long)(kt + 1) * BK * b_rs,
                                     b_cs, b_rs, tid, br);
        }
        // soft prefetch of tile kt+2 into L2 (global_prefetch_b8)
        if (kt + 2 < KT) {
            __builtin_prefetch(Ab + (long long)(kt + 2) * BK * a_cs + tid, 0, 1);
            __builtin_prefetch(Bb + (long long)(kt + 2) * BK * b_rs + tid, 0, 1);
        }

        FragAB af[4], bfr[4];
#pragma unroll
        for (int i = 0; i < 4; ++i) {
            const bf16_t* base = &sA[buf][(wm * 64 + i * 16 + l16) * LDT];
            af[i].u[0] = *(const uint4*)(base + halfA);        // K 0-7 / 8-15
            af[i].u[1] = *(const uint4*)(base + 16 + halfA);   // K 16-23 / 24-31
        }
#pragma unroll
        for (int j = 0; j < 4; ++j) {
            const bf16_t* base = &sB[buf][(wn * 64 + j * 16 + l16) * LDT + halfB];
            bfr[j].u[0] = *(const uint4*)(base);
            bfr[j].u[1] = *(const uint4*)(base + 16);
        }
#pragma unroll
        for (int i = 0; i < 4; ++i)
#pragma unroll
            for (int j = 0; j < 4; ++j)
                acc[i][j] = __builtin_amdgcn_wmma_f32_16x16x32_bf16(
                    false, af[i].v, false, bfr[j].v,
                    (short)0, acc[i][j], false, false);
        buf ^= 1;
    }

    // Epilogue: VGPR r holds row r (lanes 0-15) / row r+8 (lanes 16-31),
    // column = lane%16 within each 16x16 tile.
    const int rplus = (lane >> 4) * 8;
    const long long rowb = m0 + wm * 64;
    const long long colb = n0 + wn * 64 + l16;
#pragma unroll
    for (int i = 0; i < 4; ++i)
#pragma unroll
        for (int j = 0; j < 4; ++j) {
            const long long col = colb + j * 16;
            const float bv = bias ? bias[col] : 0.0f;
#pragma unroll
            for (int r = 0; r < 8; ++r) {
                const long long row = rowb + i * 16 + rplus + r;
                const float out = acc[i][j][r] * alpha + bv;
                const long long idx = (long long)bz * batchC + row * (long long)ldc + col;
                if constexpr (OUT_BF16) ((bf16_t*)C)[idx] = (bf16_t)out;
                else                    ((float*)C)[idx]  = out;
            }
        }
}

// --- row softmax: 8192 rows x 1024 cols, one 256-thread block per row -------
__global__ __launch_bounds__(256)
void softmax_rows(const float* __restrict__ S, bf16_t* __restrict__ P, int ncol) {
    __shared__ float red[256];
    const long long row = blockIdx.x;
    const float* s = S + row * ncol;
    const int tid = threadIdx.x;

    float lmax = -3.402823466e38f;
    for (int c = tid; c < ncol; c += 256) lmax = fmaxf(lmax, s[c]);
    red[tid] = lmax; __syncthreads();
    for (int off = 128; off > 0; off >>= 1) {
        if (tid < off) red[tid] = fmaxf(red[tid], red[tid + off]);
        __syncthreads();
    }
    const float rmax = red[0];
    __syncthreads();

    float lsum = 0.0f;
    for (int c = tid; c < ncol; c += 256) lsum += __expf(s[c] - rmax);
    red[tid] = lsum; __syncthreads();
    for (int off = 128; off > 0; off >>= 1) {
        if (tid < off) red[tid] += red[tid + off];
        __syncthreads();
    }
    const float inv = 1.0f / red[0];

    for (int c = tid; c < ncol; c += 256)
        P[row * ncol + c] = (bf16_t)(__expf(s[c] - rmax) * inv);
}

// --- host-side orchestration ------------------------------------------------
extern "C" void kernel_launch(void* const* d_in, const int* in_sizes, int n_in,
                              void* d_out, int out_size, void* d_ws, size_t ws_size,
                              hipStream_t stream) {
    (void)in_sizes; (void)n_in; (void)out_size; (void)ws_size;
    constexpr int Bsz = 8, S = 2048, Dh = 512, D = 1024;
    constexpr long long MS = (long long)Bsz * S;      // 16384
    const float inv_scale = 0.125f;                    // 1/sqrt(1024/16)

    const float* query = (const float*)d_in[0];
    const float* key   = (const float*)d_in[1];
    const float* value = (const float*)d_in[2];
    const float* Wq = (const float*)d_in[3];
    const float* bq = (const float*)d_in[4];
    const float* Wk = (const float*)d_in[5];
    const float* bk = (const float*)d_in[6];
    const float* Wv = (const float*)d_in[7];
    const float* bv = (const float*)d_in[8];
    const float* Wo = (const float*)d_in[9];
    const float* bo = (const float*)d_in[10];
    float* out = (float*)d_out;

    // Workspace: 3x bf16 [16384,1024] + 1x f32 [8,1024,1024] = 128 MiB.
    bf16_t* qb = (bf16_t*)d_ws;
    bf16_t* kb = qb + MS * D;
    bf16_t* vb = kb + MS * D;
    float*  sc = (float*)(vb + MS * D);
    bf16_t* pb = qb;   // probs reuses q (dead after scores GEMM)
    bf16_t* ab = kb;   // attn  reuses k (dead after scores GEMM)

    const dim3 blk(256);

    // q = query @ Wq^T + bq   [16384,512]x[512,1024] -> bf16
    gemm_wmma<float, float, true, true, true>
        <<<dim3(D / BN, MS / BM, 1), blk, 0, stream>>>(
            query, Wq, qb, bq, Dh,
            Dh, 1,          // A: row-major [M,K]
            1, Dh,          // B[k][n] = Wq[n*Dh+k]
            0, 0, 0, D, 1.0f);

    // k = key @ Wk^T + bk
    gemm_wmma<float, float, true, true, true>
        <<<dim3(D / BN, MS / BM, 1), blk, 0, stream>>>(
            key, Wk, kb, bk, D,
            D, 1, 1, D, 0, 0, 0, D, 1.0f);

    // v = value @ Wv^T + bv
    gemm_wmma<float, float, true, true, true>
        <<<dim3(D / BN, MS / BM, 1), blk, 0, stream>>>(
            value, Wv, vb, bv, D,
            D, 1, 1, D, 0, 0, 0, D, 1.0f);

    // scores[b] = (Q[b]^T K[b]) / 8 : A[i,s]=q[b,s,i], B[s,j]=k[b,s,j] -> f32
    gemm_wmma<bf16_t, bf16_t, false, false, false>
        <<<dim3(D / BN, D / BM, Bsz), blk, 0, stream>>>(
            qb, kb, sc, nullptr, S,
            1, D,            // A: m unit-stride, k strided by D
            D, 1,            // B: k strided by D, n unit-stride
            (long long)S * D, (long long)S * D, (long long)D * D, D, inv_scale);

    // probs = softmax(scores, axis=-1) -> bf16
    softmax_rows<<<dim3(Bsz * D), blk, 0, stream>>>(sc, pb, D);

    // attn[b] = V[b] @ P[b]^T : A[s,j]=v[b,s,j], B[j,i]=probs[b,i,j] -> bf16
    gemm_wmma<bf16_t, bf16_t, true, true, true>
        <<<dim3(D / BN, S / BM, Bsz), blk, 0, stream>>>(
            vb, pb, ab, nullptr, D,
            D, 1,            // A row-major [S,D]
            1, D,            // B[k=j][n=i] = probs[i*D+j]
            (long long)S * D, (long long)D * D, (long long)S * D, D, 1.0f);

    // out = attn @ Wo^T + bo -> f32 d_out
    gemm_wmma<bf16_t, float, true, true, false>
        <<<dim3(D / BN, MS / BM, 1), blk, 0, stream>>>(
            ab, Wo, out, bo, D,
            D, 1, 1, D, 0, 0, 0, D, 1.0f);
}
